// MambaBlock_6811818131887
// MI455X (gfx1250) — matrix operand order repaired and checked by reading, not compiled, not run on hardware
//
#include <hip/hip_runtime.h>
#include <hip/hip_bf16.h>

typedef __attribute__((ext_vector_type(16))) _Float16 v16h;
typedef __attribute__((ext_vector_type(8)))  _Float16 v8h;
typedef __attribute__((ext_vector_type(4)))  _Float16 v4h;
typedef __attribute__((ext_vector_type(8)))  float    v8f;

#define B_  8
#define L_  2048
#define DM  512
#define E_  64
#define H_  64
#define ML  (B_*L_)   // 16384

union V16 { v16h v; v8h h[2]; };

// ---- WMMA fragment loaders (contiguous k = 16h + s in A and B, 16-bit, wave32) ----
// A 16x32 f16 tile from an LDS tile with row-stride `lda` halves.
__device__ __forceinline__ v16h load_fragA_lds(const _Float16* lds, int lda, int m0, int k0) {
    int lane = threadIdx.x & 31;
    int row  = m0 + (lane & 15);
    int ksel = (lane >> 4) * 16;             // lanes 0-15: K 0..15 ; 16-31: K 16..31
    const _Float16* p = lds + row * lda + k0 + ksel;
    V16 u;
    u.h[0] = *(const v8h*)(p);               // K = ksel .. ksel+7
    u.h[1] = *(const v8h*)(p + 8);           // K = ksel+8 .. ksel+15
    return u.v;
}
// B 32x16 f16 tile from a pre-transposed [N][K] f16 weight in global memory.
__device__ __forceinline__ v16h load_fragB_glb(const _Float16* bt, int ldb, int n0, int k0) {
    int lane = threadIdx.x & 31;
    int col  = n0 + (lane & 15);
    int ks   = (lane >> 4) * 16;             // lanes 0-15: K 0..15 ; 16-31: K 16..31
    return *(const v16h*)(bt + (size_t)col * ldb + k0 + ks);   // 32B -> two b128 loads
}
__device__ __forceinline__ v8f wmma_f16(v16h a, v16h b, v8f c) {
    return __builtin_amdgcn_wmma_f32_16x16x32_f16(false, a, false, b, (short)0, c, false, false);
}
__device__ __forceinline__ v8h cvt8(float4 a, float4 b) {
    v8h h;
    h[0] = (_Float16)a.x; h[1] = (_Float16)a.y; h[2] = (_Float16)a.z; h[3] = (_Float16)a.w;
    h[4] = (_Float16)b.x; h[5] = (_Float16)b.y; h[6] = (_Float16)b.z; h[7] = (_Float16)b.w;
    return h;
}

// -------- prep: transpose/convert weights, fold W_x[:,:4]@W_dt, A=-exp(A_log) --------
__global__ __launch_bounds__(256) void k_prep(const float* __restrict__ W_in,
                                              const float* __restrict__ W_x,
                                              const float* __restrict__ W_dt,
                                              const float* __restrict__ A_log,
                                              const float* __restrict__ W_out,
                                              _Float16* __restrict__ WinT,   // [128][512]
                                              _Float16* __restrict__ WcT,    // [192][64]
                                              _Float16* __restrict__ WoutT,  // [512][64]
                                              float* __restrict__ Afac) {    // [64][64]
    const int N0 = 128 * 512, N1 = 192 * 64, N2 = 512 * 64, N3 = 64 * 64;
    const int total = N0 + N1 + N2 + N3;
    for (int i = blockIdx.x * blockDim.x + threadIdx.x; i < total; i += gridDim.x * blockDim.x) {
        if (i < N0) {
            int n = i >> 9, k = i & 511;
            WinT[i] = (_Float16)W_in[k * 128 + n];
        } else if (i < N0 + N1) {
            int j2 = i - N0; int j = j2 >> 6; int k = j2 & 63;
            float v;
            if (j < 64) {                    // folded delta weight: (W_x[:, :4] @ W_dt)[k][j]
                v = 0.f;
                #pragma unroll
                for (int r = 0; r < 4; ++r) v += W_x[k * 132 + r] * W_dt[r * 64 + j];
            } else if (j < 128) {
                v = W_x[k * 132 + 4 + (j - 64)];     // B columns
            } else {
                v = W_x[k * 132 + 68 + (j - 128)];   // C columns
            }
            WcT[j2] = (_Float16)v;
        } else if (i < N0 + N1 + N2) {
            int j2 = i - N0 - N1; int n = j2 >> 6; int k = j2 & 63;
            WoutT[j2] = (_Float16)W_out[k * 512 + n];
        } else {
            int j2 = i - N0 - N1 - N2;
            Afac[j2] = -__expf(A_log[j2]);
        }
    }
}

// -------- GEMM1: xz = x @ W_in  (M=16384,K=512,N=128), split xp_pre / res --------
__global__ __launch_bounds__(256) void k_gemm1(const float* __restrict__ x,
                                               const _Float16* __restrict__ WinT,
                                               float* __restrict__ xp_pre,
                                               float* __restrict__ res) {
    __shared__ alignas(32) _Float16 As[64 * 64];
    const int m0  = blockIdx.x * 64;
    const int tid = threadIdx.x;
    const int w   = tid >> 5, lane = tid & 31;
    const int mw  = (w & 3) * 16;
    const int nw  = (w >> 2) * 64;           // waves 0-3 -> cols 0..63, 4-7 -> 64..127
    const int sr  = tid >> 2;                // staging: row, 16 cols per thread
    const int sc  = (tid & 3) * 16;
    v8f acc[4] = {};
    for (int k0 = 0; k0 < DM; k0 += 64) {
        __syncthreads();
        {   // vectorized stage: 4x global_load_b128 -> cvt -> 2x ds_store_b128
            const float4* src = (const float4*)(x + (size_t)(m0 + sr) * DM + k0 + sc);
            float4 f0 = src[0], f1 = src[1], f2 = src[2], f3 = src[3];
            *(v8h*)(As + sr * 64 + sc)     = cvt8(f0, f1);
            *(v8h*)(As + sr * 64 + sc + 8) = cvt8(f2, f3);
        }
        __syncthreads();
        #pragma unroll
        for (int kk = 0; kk < 64; kk += 32) {
            v16h a = load_fragA_lds(As, 64, mw, kk);
            #pragma unroll
            for (int t = 0; t < 4; ++t) {
                v16h b = load_fragB_glb(WinT, DM, nw + t * 16, k0 + kk);
                acc[t] = wmma_f16(a, b, acc[t]);
            }
        }
    }
    const int rbase = m0 + mw + (lane >> 4) * 8;
    #pragma unroll
    for (int t = 0; t < 4; ++t) {
        int c = nw + t * 16 + (lane & 15);
        float* dst = (c < 64) ? (xp_pre + c) : (res + (c - 64));
        #pragma unroll
        for (int r = 0; r < 8; ++r) dst[(size_t)(rbase + r) * 64] = acc[t][r];
    }
}

// ---- conv(K=3)+SiLU fused with GEMM2: xs @ [W_delta | W_B | W_C] (K=64, N=192) ----
__global__ __launch_bounds__(256) void k_conv_gemm2(const float* __restrict__ xp_pre,
                                                    const float* __restrict__ conv_w,
                                                    const float* __restrict__ conv_b,
                                                    const _Float16* __restrict__ WcT,
                                                    const float* __restrict__ b_dt,
                                                    float* __restrict__ xps,
                                                    float* __restrict__ delta,
                                                    float* __restrict__ Bm,
                                                    float* __restrict__ Cm) {
    __shared__ alignas(32) _Float16 Xs[64 * 64];
    const int m0  = blockIdx.x * 64;
    const int l0  = m0 & (L_ - 1);           // 64-row blocks never cross a batch boundary
    const int tid = threadIdx.x;
    {   // conv: each thread does 4 rows x 4 channels with a sliding 3-tap window
        const int eg = (tid & 15) * 4;
        const int r0 = (tid >> 4) * 4;
        float4 cb = *(const float4*)(conv_b + eg);
        float wj[3][4];
        #pragma unroll
        for (int e = 0; e < 4; ++e)
            #pragma unroll
            for (int j = 0; j < 3; ++j) wj[j][e] = conv_w[(eg + e) * 3 + j];
        auto loadrow = [&](int rl) -> float4 {
            if (l0 + rl < 0) return float4{0.f, 0.f, 0.f, 0.f};
            return *(const float4*)(xp_pre + (size_t)(m0 + rl) * 64 + eg);
        };
        float4 xm2 = loadrow(r0 - 2);
        float4 xm1 = loadrow(r0 - 1);
        #pragma unroll
        for (int rr = 0; rr < 4; ++rr) {
            float4 xc = loadrow(r0 + rr);
            float4 a;
            a.x = cb.x + xm2.x * wj[0][0] + xm1.x * wj[1][0] + xc.x * wj[2][0];
            a.y = cb.y + xm2.y * wj[0][1] + xm1.y * wj[1][1] + xc.y * wj[2][1];
            a.z = cb.z + xm2.z * wj[0][2] + xm1.z * wj[1][2] + xc.z * wj[2][2];
            a.w = cb.w + xm2.w * wj[0][3] + xm1.w * wj[1][3] + xc.w * wj[2][3];
            a.x *= 1.f / (1.f + __expf(-a.x));           // SiLU
            a.y *= 1.f / (1.f + __expf(-a.y));
            a.z *= 1.f / (1.f + __expf(-a.z));
            a.w *= 1.f / (1.f + __expf(-a.w));
            *(float4*)(xps + (size_t)(m0 + r0 + rr) * 64 + eg) = a;
            v4h h; h[0] = (_Float16)a.x; h[1] = (_Float16)a.y;
                   h[2] = (_Float16)a.z; h[3] = (_Float16)a.w;
            *(v4h*)(Xs + (r0 + rr) * 64 + eg) = h;
            xm2 = xm1; xm1 = xc;
        }
    }
    __syncthreads();
    const int w = tid >> 5, lane = tid & 31;
    const int mw = (w & 3) * 16;
    const int nw = (w >> 2) * 96;            // waves 0-3 -> cols 0..95, 4-7 -> 96..191
    v8f acc[6] = {};
    #pragma unroll
    for (int k0 = 0; k0 < 64; k0 += 32) {
        v16h a = load_fragA_lds(Xs, 64, mw, k0);
        #pragma unroll
        for (int t = 0; t < 6; ++t) {
            v16h b = load_fragB_glb(WcT, 64, nw + t * 16, k0);
            acc[t] = wmma_f16(a, b, acc[t]);
        }
    }
    const int rbase = m0 + mw + (lane >> 4) * 8;
    #pragma unroll
    for (int t = 0; t < 6; ++t) {
        int c = nw + t * 16 + (lane & 15);
        if (c < 64) {
            float bd = b_dt[c];
            #pragma unroll
            for (int r = 0; r < 8; ++r) {
                float v = acc[t][r] + bd;
                delta[(size_t)(rbase + r) * 64 + c] = (v > 20.f) ? v : log1pf(__expf(v));
            }
        } else if (c < 128) {
            #pragma unroll
            for (int r = 0; r < 8; ++r) Bm[(size_t)(rbase + r) * 64 + (c - 64)] = acc[t][r];
        } else {
            #pragma unroll
            for (int r = 0; r < 8; ++r) Cm[(size_t)(rbase + r) * 64 + (c - 128)] = acc[t][r];
        }
    }
}

// -------- selective scan: one wave32 per (b,d); lane owns n = 2*lane, 2*lane+1 --------
__global__ __launch_bounds__(256) void k_scan(const float* __restrict__ delta,
                                              const float* __restrict__ Bm,
                                              const float* __restrict__ Cm,
                                              const float* __restrict__ xps,
                                              const float* __restrict__ res,
                                              const float* __restrict__ Afac,
                                              const float* __restrict__ Dv,
                                              float* __restrict__ yg) {
    const int wv   = blockIdx.x * 8 + (threadIdx.x >> 5);
    const int lane = threadIdx.x & 31;
    const int b = wv >> 6, d = wv & 63;
    const size_t base = (size_t)b * L_ * 64;
    const float* dP = delta + base + d;
    const float* xP = xps   + base + d;
    const float* rP = res   + base + d;
    const float* bP = Bm    + base + 2 * lane;
    const float* cP = Cm    + base + 2 * lane;
    float*       yP = yg    + base + d;
    const float2 Av = *(const float2*)(Afac + d * 64 + 2 * lane);
    const float  Dd = Dv[d];
    float s0 = 0.f, s1 = 0.f;
    for (int t0 = 0; t0 < L_; t0 += 32) {
        const int ob = t0 * 64;
        const int ol = ob + lane * 64;
        // chunked wave-uniform operands: 3 loads per 32 steps, broadcast via readlane
        float dch = dP[ol];
        float xch = xP[ol];
        float rch = rP[ol];
        if (t0 + 32 < L_) {   // stream the next 8KB chunk of Bm/Cm (global_prefetch_b8)
            __builtin_prefetch(bP + ob + 32 * 64 + lane * 64, 0, 1);
            __builtin_prefetch(cP + ob + 32 * 64 + lane * 64, 0, 1);
        }
        float yv = 0.f;
        #pragma unroll
        for (int s = 0; s < 32; ++s) {
            const int o = ob + s * 64;
            float dt = __shfl(dch, s, 32);
            float xv = __shfl(xch, s, 32);
            float2 bv = *(const float2*)(bP + o);     // one b64 load, coalesced
            float2 cv = *(const float2*)(cP + o);
            float du = dt * xv;
            s0 = __expf(dt * Av.x) * s0 + du * bv.x;
            s1 = __expf(dt * Av.y) * s1 + du * bv.y;
            float p = s0 * cv.x + s1 * cv.y;
            #pragma unroll
            for (int off = 16; off > 0; off >>= 1) p += __shfl_xor(p, off, 32);
            if (lane == s) yv = p;                    // lane s keeps step t0+s's result
        }
        // fused +x*D and sigmoid gate; one store per lane per chunk
        yP[ol] = (yv + xch * Dd) * (1.f / (1.f + __expf(-rch)));
    }
}

// -------- GEMM3: out = y_gated @ W_out  (M=16384,K=64,N=512) --------
__global__ __launch_bounds__(256) void k_gemm3(const float* __restrict__ yg,
                                               const _Float16* __restrict__ WoutT,
                                               float* __restrict__ out) {
    __shared__ alignas(32) _Float16 Ys[64 * 64];
    const int m0 = blockIdx.x * 64;
    const int nb = blockIdx.y * 128;
    const int tid = threadIdx.x;
    {   // vectorized stage
        const int sr = tid >> 2, sc = (tid & 3) * 16;
        const float4* src = (const float4*)(yg + (size_t)(m0 + sr) * 64 + sc);
        float4 f0 = src[0], f1 = src[1], f2 = src[2], f3 = src[3];
        *(v8h*)(Ys + sr * 64 + sc)     = cvt8(f0, f1);
        *(v8h*)(Ys + sr * 64 + sc + 8) = cvt8(f2, f3);
    }
    __syncthreads();
    const int w = tid >> 5, lane = tid & 31;
    const int mw = (w & 3) * 16;
    const int nw = nb + (w >> 2) * 64;
    v8f acc[4] = {};
    #pragma unroll
    for (int k0 = 0; k0 < 64; k0 += 32) {
        v16h a = load_fragA_lds(Ys, 64, mw, k0);
        #pragma unroll
        for (int t = 0; t < 4; ++t) {
            v16h b = load_fragB_glb(WoutT, 64, nw + t * 16, k0);
            acc[t] = wmma_f16(a, b, acc[t]);
        }
    }
    const int rbase = m0 + mw + (lane >> 4) * 8;
    #pragma unroll
    for (int t = 0; t < 4; ++t) {
        int c = nw + t * 16 + (lane & 15);
        #pragma unroll
        for (int r = 0; r < 8; ++r) out[(size_t)(rbase + r) * DM + c] = acc[t][r];
    }
}

extern "C" void kernel_launch(void* const* d_in, const int* in_sizes, int n_in,
                              void* d_out, int out_size, void* d_ws, size_t ws_size,
                              hipStream_t stream) {
    const float* x      = (const float*)d_in[0];
    const float* W_in   = (const float*)d_in[1];
    const float* conv_w = (const float*)d_in[2];
    const float* conv_b = (const float*)d_in[3];
    const float* W_x    = (const float*)d_in[4];
    const float* W_dt   = (const float*)d_in[5];
    const float* b_dt   = (const float*)d_in[6];
    const float* A_log  = (const float*)d_in[7];
    const float* Dv     = (const float*)d_in[8];
    const float* W_out  = (const float*)d_in[9];
    float* out = (float*)d_out;

    // workspace layout
    const size_t NF = (size_t)ML * 64;          // 1M floats per activation plane
    char* ws = (char*)d_ws;
    float* xp_pre = (float*)(ws);                         // 4 MB
    float* res    = (float*)(ws + 1 * NF * 4);            // 4 MB
    float* xps    = (float*)(ws + 2 * NF * 4);            // 4 MB
    float* delta  = (float*)(ws + 3 * NF * 4);            // 4 MB
    float* Bm     = (float*)(ws + 4 * NF * 4);            // 4 MB
    float* Cm     = (float*)(ws + 5 * NF * 4);            // 4 MB
    float* yg     = (float*)(ws + 6 * NF * 4);            // 4 MB
    char* wts = ws + 7 * NF * 4;
    _Float16* WinT  = (_Float16*)(wts);                   // 128*512*2 = 128 KB
    _Float16* WcT   = (_Float16*)(wts + 131072);          // 192*64*2  = 24 KB
    _Float16* WoutT = (_Float16*)(wts + 131072 + 24576);  // 512*64*2  = 64 KB
    float*    Afac  = (float*)(wts + 131072 + 24576 + 65536); // 16 KB
    const size_t need = 7 * NF * 4 + 131072 + 24576 + 65536 + 16384;
    if (ws_size < need) return;

    k_prep<<<dim3(448), dim3(256), 0, stream>>>(W_in, W_x, W_dt, A_log, W_out,
                                                WinT, WcT, WoutT, Afac);
    k_gemm1<<<dim3(ML / 64), dim3(256), 0, stream>>>(x, WinT, xp_pre, res);
    k_conv_gemm2<<<dim3(ML / 64), dim3(256), 0, stream>>>(xp_pre, conv_w, conv_b, WcT, b_dt,
                                                          xps, delta, Bm, Cm);
    k_scan<<<dim3(64), dim3(256), 0, stream>>>(delta, Bm, Cm, xps, res, Afac, Dv, yg);
    k_gemm3<<<dim3(ML / 64, 4), dim3(256), 0, stream>>>(yg, WoutT, out);
}
